// SaliencyGraphEncoder_13546326851792
// MI455X (gfx1250) — matrix operand order repaired and checked by reading, not compiled, not run on hardware
//
#include <hip/hip_runtime.h>
#include <hip/hip_fp16.h>

// ---- problem constants (match reference) ----
#define D      256
#define S      2048
#define B      8
#define F      8192
#define K1DIM  768     // 3*D
#define N1DIM  512     // 2*D
#define N2DIM  256     // D

typedef __attribute__((ext_vector_type(16))) _Float16 v16h;
typedef __attribute__((ext_vector_type(8)))  float    v8f;
typedef __attribute__((ext_vector_type(4)))  unsigned int u32x4;
typedef __attribute__((ext_vector_type(8)))  int      i32x8;
typedef __attribute__((ext_vector_type(4)))  int      i32x4;

union F16Frag { v16h v; uint4 u[2]; };
union F32Acc  { v8f  v; float f[8]; };

#if defined(__has_builtin)
#if __has_builtin(__builtin_amdgcn_tensor_load_to_lds) && __has_builtin(__builtin_amdgcn_s_wait_tensorcnt)
#define USE_TDM 1
#endif
#endif
#ifndef USE_TDM
#define USE_TDM 0
#endif

__device__ __forceinline__ float gelu_exact(float x) {
  return 0.5f * x * (1.0f + erff(x * 0.70710678118654752f));
}

#if USE_TDM
// Issue one TDM copy of a contiguous slab of `nelems` f16 elements from
// global memory into LDS. Descriptor per CDNA5 ISA.
// Flat-address low 32 bits of a __shared__ pointer are the LDS byte offset
// (LDS aperture rule).
__device__ __forceinline__ void tdm_load_slab(const _Float16* gsrc, void* ldst,
                                              int nelems) {
  unsigned lds_addr = (unsigned)(unsigned long long)ldst;
  unsigned long long ga = (unsigned long long)gsrc;

  u32x4 g0;
  g0.x = 1u;                                             // count=1 (valid user D#)
  g0.y = lds_addr;                                       // lds_addr [63:32]
  g0.z = (unsigned)(ga & 0xffffffffu);                   // global_addr lo
  g0.w = (unsigned)((ga >> 32) & 0x01ffffffu) | (2u << 30); // addr hi + type=2

  i32x8 g1;
  g1[0] = 0x00010000;                                    // data_size=1 (2 bytes)
  g1[1] = (int)((nelems & 0xffff) << 16);                // tensor_dim0 lo16
  g1[2] = (int)(((unsigned)nelems >> 16) & 0xffffu) | (1 << 16); // dim0 hi | dim1=1
  g1[3] = (int)((nelems & 0xffff) << 16);                // tile_dim0 = nelems
  g1[4] = 1;                                             // tile_dim1 = 1
  g1[5] = nelems;                                        // tensor_dim0_stride lo32
  g1[6] = 0;
  g1[7] = 0;

  i32x4 gz4 = {0, 0, 0, 0};
#if __clang_major__ >= 23
  i32x8 gz8 = {0, 0, 0, 0, 0, 0, 0, 0};
  __builtin_amdgcn_tensor_load_to_lds(g0, g1, gz4, gz4, gz8, 0);
#else
  __builtin_amdgcn_tensor_load_to_lds(g0, g1, gz4, gz4, 0);
#endif
}
#endif

// ---------------------------------------------------------------------------
// Kernel 0: zero agg/pooled; build fp16 weights, pre-transposed AND slab-major:
//   w1ts[((kk*512)+n)*32 + k] = w1[(kk*32+k)*512 + n]   (24 slabs of 32KB)
//   w2ts[((kk*256)+n)*32 + k] = w2[(kk*32+k)*256 + n]   (16 slabs of 16KB)
// so each K-slab the GEMM needs is one contiguous block -> ideal for TDM.
// ---------------------------------------------------------------------------
__global__ __launch_bounds__(256)
void prep_kernel(const float* __restrict__ w1, const float* __restrict__ w2,
                 _Float16* __restrict__ w1ts, _Float16* __restrict__ w2ts,
                 float* __restrict__ agg, float* __restrict__ pooled)
{
  size_t i = (size_t)blockIdx.x * 256 + threadIdx.x;
  if (i < (size_t)B * S * D) agg[i] = 0.0f;
  if (i < (size_t)K1DIM * N1DIM) {
    int kk = (int)(i >> 14);          // / 16384
    int r  = (int)(i & 16383);
    int n  = r >> 5;
    int k  = r & 31;
    w1ts[i] = (_Float16)w1[(size_t)(kk * 32 + k) * N1DIM + n];
  }
  if (i < (size_t)N1DIM * N2DIM) {
    int kk = (int)(i >> 13);          // / 8192
    int r  = (int)(i & 8191);
    int n  = r >> 5;
    int k  = r & 31;
    w2ts[i] = (_Float16)w2[(size_t)(kk * 32 + k) * N2DIM + n];
  }
  if (i < (size_t)B * D) pooled[i] = 0.0f;
}

// ---------------------------------------------------------------------------
// Kernel 1: fused edge MLP (WMMA f32_16x16x32_f16) + atomic scatter into agg
// One block = 16 edge-direction rows, 256 threads = 8 wave32s.
// Weight slabs are DMA'd into LDS by the Tensor Data Mover (wave 0), waves
// consume them via 16B ds_load fragments feeding v_wmma.
// ---------------------------------------------------------------------------
__global__ __launch_bounds__(256)
void edge_mlp_kernel(const float* __restrict__ pos_emb,
                     const float* __restrict__ pred_emb,
                     const float* __restrict__ role_emb,
                     const _Float16* __restrict__ w1ts,
                     const float* __restrict__ b1,
                     const _Float16* __restrict__ w2ts,
                     const float* __restrict__ b2,
                     const int* __restrict__ pred_idx,
                     const int* __restrict__ a0,
                     const int* __restrict__ a1,
                     const int* __restrict__ role_idx,
                     const int* __restrict__ is_role,
                     const int* __restrict__ add_rev,
                     const int* __restrict__ maskp,
                     float* __restrict__ agg)
{
  // A tile (16x768 f16 = 24KB); reused as H tile (16x512 f16) for layer 2
  __shared__ __align__(16) _Float16 shA[16][K1DIM];
  // transposed weight slab Wt[n][k] (512x32 f16 = 32KB); layer2 uses rows 0..255
  __shared__ __align__(16) _Float16 shWt[N1DIM][32];
  __shared__ unsigned long long rp0[16], rp1[16], rp2[16];
  __shared__ float rw[16];
  __shared__ int   rtgt[16], rb[16];

  const int tid  = threadIdx.x;
  const int lane = tid & 31;
  const int wave = tid >> 5;

  // ---- per-row metadata: gather sources, weight, scatter target ----
  if (tid < 16) {
    int g   = blockIdx.x * 16 + tid;   // edge-direction index
    int e   = g >> 1;                  // edge index in [0, B*F)
    int dir = g & 1;                   // 0 = fwd, 1 = rev
    int bb  = e >> 13;                 // e / F
    int pi = pred_idx[e], i0 = a0[e], i1 = a1[e], ri = role_idx[e];
    int ro = is_role[e], ar = add_rev[e], mk = maskp[e];
    const float* n0 = pos_emb + (size_t)i0 * D;
    const float* n1 = pos_emb + (size_t)i1 * D;
    const float* pe = pred_emb + (size_t)pi * D;
    const float* re = role_emb + (size_t)ri * D;
    const float* p0; const float* p2; float w; int tgt;
    if (dir == 0) {                    // x_fwd = [n0, pe, role? re : n1]
      p0 = n0; p2 = ro ? re : n1;
      w = mk ? 1.0f : 0.0f;
      tgt = ro ? i0 : i1;
    } else {                           // x_rev = [n1, pe, n0]
      p0 = n1; p2 = n0;
      w = (mk && !ro && ar) ? 1.0f : 0.0f;
      tgt = i0;
    }
    rp0[tid] = (unsigned long long)p0;
    rp1[tid] = (unsigned long long)pe;
    rp2[tid] = (unsigned long long)p2;
    rw[tid] = w; rtgt[tid] = tgt; rb[tid] = bb;
  }
  __syncthreads();

  // ---- stage A tile: gather fp32 rows, convert to fp16 in LDS ----
  {
    int row = tid >> 4;
    int sub = tid & 15;                // 16 threads per row, 16 floats per seg
    const float* s0 = (const float*)rp0[row];
    const float* s1 = (const float*)rp1[row];
    const float* s2 = (const float*)rp2[row];
    const float* srcs[3] = { s0, s1, s2 };
    #pragma unroll
    for (int seg = 0; seg < 3; ++seg) {
      const float* sp = srcs[seg] + sub * 16;
      _Float16* dst = &shA[row][seg * D + sub * 16];
      #pragma unroll
      for (int q = 0; q < 4; ++q) {
        float4 v = reinterpret_cast<const float4*>(sp)[q];
        dst[q * 4 + 0] = (_Float16)v.x; dst[q * 4 + 1] = (_Float16)v.y;
        dst[q * 4 + 2] = (_Float16)v.z; dst[q * 4 + 3] = (_Float16)v.w;
      }
    }
  }

  // WMMA fragment addressing (wave32 layouts from CDNA5 ISA)
  const int mA  = lane & 15;                 // A-matrix row
  const int kbA = (lane < 16) ? 0 : 8;       // A frag: K blocks {kbA..+7, kbA+16..+23}
  const int kbB = (lane < 16) ? 0 : 8;       // B frag: K blocks {kbB..+7, kbB+16..+23}
  const int nB  = lane & 15;                 // B/C/D column within tile

  // ---- layer 1: 16x512 = A(16x768) * W1(768x512), K-slabs of 32 ----
  F32Acc acc1[4];
  #pragma unroll
  for (int s = 0; s < 4; ++s)
    #pragma unroll
    for (int q = 0; q < 8; ++q) acc1[s].f[q] = 0.0f;

  for (int kk = 0; kk < K1DIM / 32; ++kk) {
    // stage weight slab (contiguous 32KB) into LDS
#if USE_TDM
    if (wave == 0) {
      tdm_load_slab(w1ts + (size_t)kk * 32 * N1DIM, &shWt[0][0], 32 * N1DIM);
      __builtin_amdgcn_s_wait_tensorcnt(0);
    }
#else
    {
      const uint4* src = reinterpret_cast<const uint4*>(w1ts + (size_t)kk * 32 * N1DIM);
      uint4* dst = reinterpret_cast<uint4*>(&shWt[0][0]);
      for (int i = tid; i < (32 * N1DIM) / 8; i += 256) dst[i] = src[i];
    }
#endif
    __syncthreads();

    F16Frag afr;
    afr.u[0] = *reinterpret_cast<const uint4*>(&shA[mA][kk * 32 + kbA]);
    afr.u[1] = *reinterpret_cast<const uint4*>(&shA[mA][kk * 32 + kbA + 16]);
    #pragma unroll
    for (int s = 0; s < 4; ++s) {
      int n = wave * 64 + s * 16 + nB;
      F16Frag bfr;
      bfr.u[0] = *reinterpret_cast<const uint4*>(&shWt[n][kbB]);
      bfr.u[1] = *reinterpret_cast<const uint4*>(&shWt[n][kbB + 16]);
      acc1[s].v = __builtin_amdgcn_wmma_f32_16x16x32_f16(
          false, afr.v, false, bfr.v, (short)0, acc1[s].v, false, false);
    }
    __syncthreads();
  }

  // ---- bias + exact GELU, write H tile (16x512 f16) over the A buffer ----
  _Float16 (*shH)[N1DIM] = reinterpret_cast<_Float16 (*)[N1DIM]>(&shA[0][0]);
  #pragma unroll
  for (int s = 0; s < 4; ++s) {
    int n = wave * 64 + s * 16 + nB;
    float bias = b1[n];
    #pragma unroll
    for (int r = 0; r < 8; ++r) {
      int m = r + ((lane < 16) ? 0 : 8);
      shH[m][n] = (_Float16)gelu_exact(acc1[s].f[r] + bias);
    }
  }
  __syncthreads();

  // ---- layer 2: 16x256 = H(16x512) * W2(512x256) ----
  F32Acc acc2[2];
  #pragma unroll
  for (int s = 0; s < 2; ++s)
    #pragma unroll
    for (int q = 0; q < 8; ++q) acc2[s].f[q] = 0.0f;

  for (int kk = 0; kk < N1DIM / 32; ++kk) {
#if USE_TDM
    if (wave == 0) {
      tdm_load_slab(w2ts + (size_t)kk * 32 * N2DIM, &shWt[0][0], 32 * N2DIM);
      __builtin_amdgcn_s_wait_tensorcnt(0);
    }
#else
    {
      const uint4* src = reinterpret_cast<const uint4*>(w2ts + (size_t)kk * 32 * N2DIM);
      uint4* dst = reinterpret_cast<uint4*>(&shWt[0][0]);
      for (int i = tid; i < (32 * N2DIM) / 8; i += 256) dst[i] = src[i];
    }
#endif
    __syncthreads();

    F16Frag afr;
    afr.u[0] = *reinterpret_cast<const uint4*>(&shH[mA][kk * 32 + kbA]);
    afr.u[1] = *reinterpret_cast<const uint4*>(&shH[mA][kk * 32 + kbA + 16]);
    #pragma unroll
    for (int s = 0; s < 2; ++s) {
      int n = wave * 32 + s * 16 + nB;
      F16Frag bfr;
      bfr.u[0] = *reinterpret_cast<const uint4*>(&shWt[n][kbB]);
      bfr.u[1] = *reinterpret_cast<const uint4*>(&shWt[n][kbB + 16]);
      acc2[s].v = __builtin_amdgcn_wmma_f32_16x16x32_f16(
          false, afr.v, false, bfr.v, (short)0, acc2[s].v, false, false);
    }
    __syncthreads();
  }

  // ---- bias + weighted atomic scatter into agg[b, tgt, :] ----
  #pragma unroll
  for (int s = 0; s < 2; ++s) {
    int n = wave * 32 + s * 16 + nB;
    float bias = b2[n];
    #pragma unroll
    for (int r = 0; r < 8; ++r) {
      int m = r + ((lane < 16) ? 0 : 8);
      float w = rw[m];
      if (w != 0.0f) {
        float val = acc2[s].f[r] + bias;   // w is 0 or 1
        size_t off = (((size_t)rb[m] << 11) + (size_t)rtgt[m]) * D + n;
        atomicAdd(&agg[off], val);
      }
    }
  }
}

// ---------------------------------------------------------------------------
// Kernel 2: x = nodes + agg ; LayerNorm ; accumulate pooled sum per batch.
// One wave32 per row (D=256 -> 8 elems/lane), 8 rows per block.
// ---------------------------------------------------------------------------
__global__ __launch_bounds__(256)
void ln_pool_kernel(const float* __restrict__ pos_emb,
                    const float* __restrict__ agg,
                    const float* __restrict__ ln_g,
                    const float* __restrict__ ln_b,
                    float* __restrict__ pooled)
{
  __shared__ float spool[D];
  const int tid  = threadIdx.x;
  const int lane = tid & 31;
  const int wave = tid >> 5;
  if (tid < D) spool[tid] = 0.0f;
  __syncthreads();

  int row = blockIdx.x * 8 + wave;       // in [0, B*S); all rows share batch b
  int s   = row & (S - 1);
  float x[8];
  float sum = 0.0f;
  #pragma unroll
  for (int j = 0; j < 8; ++j) {
    int c = j * 32 + lane;
    x[j] = pos_emb[(size_t)s * D + c] + agg[(size_t)row * D + c];
    sum += x[j];
  }
  #pragma unroll
  for (int off = 16; off > 0; off >>= 1) sum += __shfl_xor(sum, off, 32);
  float mu = sum * (1.0f / D);
  float vs = 0.0f;
  #pragma unroll
  for (int j = 0; j < 8; ++j) { float d = x[j] - mu; vs += d * d; }
  #pragma unroll
  for (int off = 16; off > 0; off >>= 1) vs += __shfl_xor(vs, off, 32);
  float inv = rsqrtf(vs * (1.0f / D) + 1e-5f);
  #pragma unroll
  for (int j = 0; j < 8; ++j) {
    int c = j * 32 + lane;
    float y = (x[j] - mu) * inv * ln_g[c] + ln_b[c];
    atomicAdd(&spool[c], y);
  }
  __syncthreads();
  if (tid < D) {
    int b = (blockIdx.x * 8) >> 11;      // row / S
    atomicAdd(&pooled[b * D + tid], spool[tid]);
  }
}

// ---------------------------------------------------------------------------
// Kernel 3: head — pooled/S -> GELU(lw1) -> lw2 -> out (8x256, tiny)
// ---------------------------------------------------------------------------
__global__ __launch_bounds__(256)
void head_kernel(const float* __restrict__ pooled,
                 const float* __restrict__ lw1, const float* __restrict__ lb1,
                 const float* __restrict__ lw2, const float* __restrict__ lb2,
                 float* __restrict__ out)
{
  __shared__ float sp[B * D];
  __shared__ float hh[B * D];
  const int tid = threadIdx.x;
  for (int i = tid; i < B * D; i += 256) sp[i] = pooled[i] * (1.0f / S);
  __syncthreads();
  for (int i = tid; i < B * D; i += 256) {
    int b = i >> 8, j = i & 255;
    float acc = lb1[j];
    for (int k = 0; k < D; ++k) acc += sp[b * D + k] * lw1[k * D + j];
    hh[i] = gelu_exact(acc);
  }
  __syncthreads();
  for (int i = tid; i < B * D; i += 256) {
    int b = i >> 8, j = i & 255;
    float acc = lb2[j];
    for (int k = 0; k < D; ++k) acc += hh[b * D + k] * lw2[k * D + j];
    out[i] = acc;
  }
}

// ---------------------------------------------------------------------------
extern "C" void kernel_launch(void* const* d_in, const int* in_sizes, int n_in,
                              void* d_out, int out_size, void* d_ws, size_t ws_size,
                              hipStream_t stream)
{
  const float* pos_emb  = (const float*)d_in[0];
  const float* pred_emb = (const float*)d_in[1];
  const float* role_emb = (const float*)d_in[2];
  const float* w1   = (const float*)d_in[3];
  const float* b1   = (const float*)d_in[4];
  const float* w2   = (const float*)d_in[5];
  const float* b2   = (const float*)d_in[6];
  const float* ln_g = (const float*)d_in[7];
  const float* ln_b = (const float*)d_in[8];
  const float* lw1  = (const float*)d_in[9];
  const float* lb1  = (const float*)d_in[10];
  const float* lw2  = (const float*)d_in[11];
  const float* lb2  = (const float*)d_in[12];
  const int* pred_idx = (const int*)d_in[13];
  const int* a0       = (const int*)d_in[14];
  const int* a1       = (const int*)d_in[15];
  const int* role_idx = (const int*)d_in[16];
  const int* is_role  = (const int*)d_in[17];
  const int* add_rev  = (const int*)d_in[18];
  const int* mask     = (const int*)d_in[19];
  float* out = (float*)d_out;

  char* ws = (char*)d_ws;
  float* agg = (float*)ws;         ws += (size_t)B * S * D * sizeof(float);
  _Float16* w1ts = (_Float16*)ws;  ws += (size_t)K1DIM * N1DIM * sizeof(_Float16);
  _Float16* w2ts = (_Float16*)ws;  ws += (size_t)N1DIM * N2DIM * sizeof(_Float16);
  float* pooled = (float*)ws;

  prep_kernel<<<(B * S * D) / 256, 256, 0, stream>>>(w1, w2, w1ts, w2ts, agg, pooled);
  edge_mlp_kernel<<<(B * F * 2) / 16, 256, 0, stream>>>(
      pos_emb, pred_emb, role_emb, w1ts, b1, w2ts, b2,
      pred_idx, a0, a1, role_idx, is_role, add_rev, mask, agg);
  ln_pool_kernel<<<(B * S) / 8, 256, 0, stream>>>(pos_emb, agg, ln_g, ln_b, pooled);
  head_kernel<<<1, 256, 0, stream>>>(pooled, lw1, lb1, lw2, lb2, out);
}
